// TOPStrategyGNN_68796786147902
// MI455X (gfx1250) — compile-verified, matrix operand order checked by reading, not run on hardware
//
#include <hip/hip_runtime.h>
#include <hip/hip_bf16.h>

typedef __attribute__((ext_vector_type(16))) _Float16 v16h;
typedef __attribute__((ext_vector_type(8)))  _Float16 v8h;
typedef __attribute__((ext_vector_type(8)))  float    v8f;

#define NN   100000
#define EE   800000
#define E2T  (NN + EE)     // edges + self loops
#define HIDC 64

static inline int cdiv(int a, int b) { return (a + b - 1) / b; }

// ---------------------------------------------------------------------------
// WMMA fragment load: 16-bit A/B fragment for v_wmma_f32_16x16x32_f16.
// Matrix stored row-major [rows x ld] fp16 (for B we pre-transpose so that
// "rows" are output columns). Per ISA layout: lanes 0-15 hold row (lane&15)
// with K = {k0..k0+7, k0+16..k0+23}; lanes 16-31 hold K = {k0+8..15, k0+24..31}.
// ---------------------------------------------------------------------------
__device__ __forceinline__ v16h load_frag16(const _Float16* __restrict__ base,
                                            int rowBase, int ld, int k0, int lane) {
  int r   = lane & 15;
  int sel = (lane >> 4) << 3;            // 0 or 8 halves
  const _Float16* p = base + (size_t)(rowBase + r) * (size_t)ld + (size_t)(k0 + sel);
  v8h lo = *(const v8h*)(p);             // K = k0+sel .. +7
  v8h hi = *(const v8h*)(p + 16);        // K = k0+sel+16 .. +23
  return __builtin_shufflevector(lo, hi, 0,1,2,3,4,5,6,7,8,9,10,11,12,13,14,15);
}

__device__ __forceinline__ void atomicMaxF(float* addr, float v) {
  if (v >= 0.0f) atomicMax((int*)addr, __float_as_int(v));
  else           atomicMin((unsigned int*)addr, __float_as_uint(v));
}

// ---------------------------------------------------------------------------
// Generic fill
// ---------------------------------------------------------------------------
__global__ void fill_kernel(float* __restrict__ p, float v, int count) {
  int i = blockIdx.x * blockDim.x + threadIdx.x;
  if (i < count) p[i] = v;
}

// ---------------------------------------------------------------------------
// fp32 -> fp16 conversions: x, and weight transposes (lin_w^T, W_in^T)
// ---------------------------------------------------------------------------
__global__ void cvt_x_kernel(const float* __restrict__ x, _Float16* __restrict__ x16) {
  int i = blockIdx.x * blockDim.x + threadIdx.x;
  if (i < NN * 32) x16[i] = (_Float16)x[i];
}

__global__ void cvt_w_kernel(const float* __restrict__ lin_w, const float* __restrict__ Win,
                             _Float16* __restrict__ wt, _Float16* __restrict__ winT) {
  int idx = blockIdx.x * blockDim.x + threadIdx.x;
  const int TW = 3 * 256 * 64;
  if (idx < TW) {
    int l = idx / (256 * 64);
    int rem = idx % (256 * 64);
    int nn = rem >> 6;            // output channel 0..255
    int k  = rem & 63;            // input channel 0..63
    wt[idx] = (_Float16)lin_w[(size_t)(l * 64 + k) * 256 + nn];   // wt[l][n][k]
  } else if (idx < TW + 64 * 32) {
    int j = idx - TW;
    int c = j >> 5;               // 0..63
    int k = j & 31;               // 0..31
    winT[j] = (_Float16)Win[k * 64 + c];                          // winT[c][k]
  }
}

// ---------------------------------------------------------------------------
// Single-block prep: ea mean, folded attention weights, folded BN params.
//   Wae[l][dd][hd] = sum_c lin_edge_w[l][dd][hd*64+c] * att_edge[l][hd][c]
//   lae[l][hd]     = sum_dd ea_mean[dd] * Wae[l][dd][hd]       (self-loop a_edge)
//   Asrc[l][k][hd] = sum_c lin_w[l][k][hd*64+c] * att_src[l][hd][c]  (Adst same)
//   isc/ish: input BN scale/shift with Linear bias folded
//   lsc/lsh: per-layer BN scale/shift with gat_bias folded
// ---------------------------------------------------------------------------
__global__ void __launch_bounds__(256) prep_kernel(
    const float* __restrict__ ea, const float* __restrict__ lin_w,
    const float* __restrict__ lin_edge_w, const float* __restrict__ att_src,
    const float* __restrict__ att_dst, const float* __restrict__ att_edge,
    const float* __restrict__ b_in, const float* __restrict__ big,
    const float* __restrict__ bib, const float* __restrict__ bim,
    const float* __restrict__ biv, const float* __restrict__ gat_bias,
    const float* __restrict__ bg, const float* __restrict__ bb,
    const float* __restrict__ bm, const float* __restrict__ bv,
    float* __restrict__ Wae, float* __restrict__ lae,
    float* __restrict__ Asrc, float* __restrict__ Adst,
    float* __restrict__ isc, float* __restrict__ ish,
    float* __restrict__ lsc, float* __restrict__ lsh) {
  __shared__ float red[256];
  __shared__ float smean[4];
  int tid = threadIdx.x;

  // edge_attr mean over EE edges, 4 dims
  float part[4] = {0.f, 0.f, 0.f, 0.f};
  for (int e = tid; e < EE; e += 256) {
    part[0] += ea[(size_t)e * 4 + 0];
    part[1] += ea[(size_t)e * 4 + 1];
    part[2] += ea[(size_t)e * 4 + 2];
    part[3] += ea[(size_t)e * 4 + 3];
  }
  for (int d = 0; d < 4; ++d) {
    red[tid] = part[d];
    __syncthreads();
    for (int s = 128; s > 0; s >>= 1) {
      if (tid < s) red[tid] += red[tid + s];
      __syncthreads();
    }
    if (tid == 0) smean[d] = red[0] / (float)EE;
    __syncthreads();
  }

  // Wae : 3*4*4
  for (int i = tid; i < 48; i += 256) {
    int l = i >> 4, dd = (i >> 2) & 3, hd = i & 3;
    const float* lw = lin_edge_w + (size_t)(l * 4 + dd) * 256 + hd * 64;
    const float* at = att_edge + (size_t)(l * 4 + hd) * 64;
    float acc = 0.f;
    for (int c = 0; c < 64; ++c) acc += lw[c] * at[c];
    Wae[(l * 4 + dd) * 4 + hd] = acc;
  }
  __syncthreads();
  // lae : 3*4
  for (int i = tid; i < 12; i += 256) {
    int l = i >> 2, hd = i & 3;
    float acc = 0.f;
    for (int dd = 0; dd < 4; ++dd) acc += smean[dd] * Wae[(l * 4 + dd) * 4 + hd];
    lae[i] = acc;
  }
  // Asrc / Adst : 3*64*4
  for (int i = tid; i < 768; i += 256) {
    int l = i / 256, k = (i >> 2) & 63, hd = i & 3;
    const float* lw = lin_w + (size_t)(l * 64 + k) * 256 + hd * 64;
    const float* s1 = att_src + (size_t)(l * 4 + hd) * 64;
    const float* s2 = att_dst + (size_t)(l * 4 + hd) * 64;
    float a1 = 0.f, a2 = 0.f;
    for (int c = 0; c < 64; ++c) { a1 += lw[c] * s1[c]; a2 += lw[c] * s2[c]; }
    Asrc[i] = a1;
    Adst[i] = a2;
  }
  // input BN (fold Linear bias)
  for (int c = tid; c < 64; c += 256) {
    float s = big[c] * rsqrtf(biv[c] + 1e-5f);
    isc[c] = s;
    ish[c] = (b_in[c] - bim[c]) * s + bib[c];
  }
  // layer BN (fold gat_bias) : 3*64
  for (int i = tid; i < 192; i += 256) {
    float s = bg[i] * rsqrtf(bv[i] + 1e-5f);
    lsc[i] = s;
    lsh[i] = (gat_bias[i] - bm[i]) * s + bb[i];
  }
}

// ---------------------------------------------------------------------------
// Input GEMM: h = relu( (x[100k,32] @ W_in[32,64]) * isc + ish ).
// One wave = one 16-row M-tile x all four 16-col N-tiles: the A fragment is
// loaded once and reused by 4 WMMAs (weights stay L0-resident).
// ---------------------------------------------------------------------------
__global__ void __launch_bounds__(256) gemm_in_kernel(
    const _Float16* __restrict__ x16, const _Float16* __restrict__ winT,
    const float* __restrict__ isc, const float* __restrict__ ish,
    float* __restrict__ h, _Float16* __restrict__ h16) {
  int lane = threadIdx.x & 31;
  int tm = blockIdx.x * 8 + (threadIdx.x >> 5);   // one M-tile per wave
  if (tm >= NN / 16) return;                      // wave-uniform
  v16h a = load_frag16(x16, tm * 16, 32, 0, lane);
  v8f c[4];
  #pragma unroll
  for (int q = 0; q < 4; ++q) {
    v8f z = {};
    v16h b = load_frag16(winT, q * 16, 32, 0, lane);
    c[q] = __builtin_amdgcn_wmma_f32_16x16x32_f16(false, a, false, b, (short)0, z, false, false);
  }
  int n  = lane & 15;
  int mb = (lane >> 4) << 3;
  #pragma unroll
  for (int q = 0; q < 4; ++q) {
    int col = q * 16 + n;
    float s = isc[col], t = ish[col];
    #pragma unroll
    for (int r = 0; r < 8; ++r) {
      int row = tm * 16 + mb + r;
      float y = fmaxf(c[q][r] * s + t, 0.f);
      h[(size_t)row * HIDC + col]   = y;
      h16[(size_t)row * HIDC + col] = (_Float16)y;
    }
  }
}

// ---------------------------------------------------------------------------
// Layer GEMM: xw[100k,256] (fp16) = h16[100k,64] @ lin_w[64,256], WMMA, K=64.
// One wave = one M-tile x four N-tiles (tq selects which quarter of the 16
// N-tiles): each A fragment loaded once, used by 4 WMMAs; 8 WMMAs per wave.
// Bt is lin_w pre-transposed: [256,64] fp16.
// ---------------------------------------------------------------------------
__global__ void __launch_bounds__(256) gemm_xw_kernel(
    const _Float16* __restrict__ hA, const _Float16* __restrict__ Bt,
    _Float16* __restrict__ xw) {
  int lane = threadIdx.x & 31;
  int wave = blockIdx.x * 8 + (threadIdx.x >> 5);
  int tm = wave >> 2;                       // 0..6249
  int tq = wave & 3;                        // N-tile group: tn = tq*4 .. tq*4+3
  if (tm >= NN / 16) return;                // wave-uniform
  v8f c[4] = {};
  #pragma unroll
  for (int k0 = 0; k0 < HIDC; k0 += 32) {
    v16h a = load_frag16(hA, tm * 16, HIDC, k0, lane);
    #pragma unroll
    for (int q = 0; q < 4; ++q) {
      v16h b = load_frag16(Bt, (tq * 4 + q) * 16, HIDC, k0, lane);
      c[q] = __builtin_amdgcn_wmma_f32_16x16x32_f16(false, a, false, b, (short)0, c[q], false, false);
    }
  }
  int n  = lane & 15;
  int mb = (lane >> 4) << 3;
  #pragma unroll
  for (int q = 0; q < 4; ++q) {
    size_t base = (size_t)(tm * 16 + mb) * 256 + (tq * 4 + q) * 16 + n;
    #pragma unroll
    for (int r = 0; r < 8; ++r) xw[base + (size_t)r * 256] = (_Float16)c[q][r];
  }
}

// ---------------------------------------------------------------------------
// Per-node attention scalars: a_src[n,h] = h[n,:] . Asrc[:,h]  (Asrc [64][4])
// ---------------------------------------------------------------------------
__global__ void node_attn_kernel(const float* __restrict__ h,
                                 const float* __restrict__ Asrc,
                                 const float* __restrict__ Adst,
                                 float* __restrict__ asrc, float* __restrict__ adst) {
  int n = blockIdx.x * blockDim.x + threadIdx.x;
  if (n >= NN) return;
  const float* hr = h + (size_t)n * HIDC;
  float as[4] = {0.f, 0.f, 0.f, 0.f};
  float ad[4] = {0.f, 0.f, 0.f, 0.f};
  for (int k = 0; k < HIDC; ++k) {
    float hv = hr[k];
    #pragma unroll
    for (int hd = 0; hd < 4; ++hd) {
      as[hd] += hv * Asrc[k * 4 + hd];
      ad[hd] += hv * Adst[k * 4 + hd];
    }
  }
  #pragma unroll
  for (int hd = 0; hd < 4; ++hd) {
    asrc[(size_t)n * 4 + hd] = as[hd];
    adst[(size_t)n * 4 + hd] = ad[hd];
  }
}

// ---------------------------------------------------------------------------
// Edge pass 1: logits (leaky relu) + segment max (atomic). e >= EE => self loop
// ---------------------------------------------------------------------------
__global__ void edge_p1_kernel(const int* __restrict__ src, const int* __restrict__ dst,
                               const float* __restrict__ ea,
                               const float* __restrict__ Wae, const float* __restrict__ lae,
                               const float* __restrict__ asrc, const float* __restrict__ adst,
                               float* __restrict__ logits, float* __restrict__ mx) {
  int e = blockIdx.x * blockDim.x + threadIdx.x;
  if (e >= E2T) return;
  int s, d;
  float aeh[4];
  if (e < EE) {
    s = src[e]; d = dst[e];
    float e0 = ea[(size_t)e * 4 + 0], e1 = ea[(size_t)e * 4 + 1];
    float e2 = ea[(size_t)e * 4 + 2], e3 = ea[(size_t)e * 4 + 3];
    #pragma unroll
    for (int hd = 0; hd < 4; ++hd)
      aeh[hd] = e0 * Wae[0 * 4 + hd] + e1 * Wae[1 * 4 + hd] +
                e2 * Wae[2 * 4 + hd] + e3 * Wae[3 * 4 + hd];
  } else {
    s = d = e - EE;
    #pragma unroll
    for (int hd = 0; hd < 4; ++hd) aeh[hd] = lae[hd];
  }
  #pragma unroll
  for (int hd = 0; hd < 4; ++hd) {
    float v = asrc[(size_t)s * 4 + hd] + adst[(size_t)d * 4 + hd] + aeh[hd];
    v = v > 0.f ? v : 0.2f * v;
    logits[(size_t)e * 4 + hd] = v;
    atomicMaxF(&mx[(size_t)d * 4 + hd], v);
  }
}

// ---------------------------------------------------------------------------
// Edge pass 2: ex = exp(logit - mx[dst]); den[dst] += ex (overwrite logits=ex)
// ---------------------------------------------------------------------------
__global__ void edge_p2_kernel(const int* __restrict__ dst,
                               const float* __restrict__ mx,
                               float* __restrict__ logits, float* __restrict__ den) {
  int e = blockIdx.x * blockDim.x + threadIdx.x;
  if (e >= E2T) return;
  int d = (e < EE) ? dst[e] : (e - EE);
  #pragma unroll
  for (int hd = 0; hd < 4; ++hd) {
    float ex = __expf(logits[(size_t)e * 4 + hd] - mx[(size_t)d * 4 + hd]);
    logits[(size_t)e * 4 + hd] = ex;
    atomicAdd(&den[(size_t)d * 4 + hd], ex);
  }
}

// ---------------------------------------------------------------------------
// Edge pass 3: one wave per edge; out[dst, c] += 0.25 * alpha[h] * xw[src, h, c]
// (head mean folded into accumulation; 256 channels over 8 lane-iterations)
// ---------------------------------------------------------------------------
__global__ void __launch_bounds__(256) edge_p3_kernel(
    const int* __restrict__ src, const int* __restrict__ dst,
    const _Float16* __restrict__ xw, const float* __restrict__ exv,
    const float* __restrict__ den, float* __restrict__ out) {
  int lane = threadIdx.x & 31;
  int e = blockIdx.x * 8 + (threadIdx.x >> 5);
  if (e >= E2T) return;                     // wave-uniform
  int s, d;
  if (e < EE) { s = src[e]; d = dst[e]; } else { s = d = e - EE; }
  float al[4];
  #pragma unroll
  for (int hd = 0; hd < 4; ++hd)
    al[hd] = exv[(size_t)e * 4 + hd] / (den[(size_t)d * 4 + hd] + 1e-16f) * 0.25f;
  const _Float16* xr = xw + (size_t)s * 256;
  float* orow = out + (size_t)d * HIDC;
  #pragma unroll
  for (int i = 0; i < 8; ++i) {
    int ch = lane + 32 * i;
    float v = (float)xr[ch] * al[ch >> 6];
    atomicAdd(&orow[ch & 63], v);
  }
}

// ---------------------------------------------------------------------------
// Layer finalize: h = relu(out*lsc + lsh) + h  (BN + bias folded); also fp16 h
// ---------------------------------------------------------------------------
__global__ void finalize_kernel(const float* __restrict__ out,
                                const float* __restrict__ lsc, const float* __restrict__ lsh,
                                float* __restrict__ h, _Float16* __restrict__ h16) {
  int idx = blockIdx.x * blockDim.x + threadIdx.x;
  if (idx >= NN * HIDC) return;
  int c = idx & 63;
  float y = fmaxf(out[idx] * lsc[c] + lsh[c], 0.f) + h[idx];
  h[idx] = y;
  h16[idx] = (_Float16)y;
}

// ---------------------------------------------------------------------------
// Output head: pred = relu(h@Wo1+bo1)@Wo2+bo2 ; blend with gap sensitivity.
// ---------------------------------------------------------------------------
__global__ void __launch_bounds__(256) head_kernel(
    const float* __restrict__ h, const float* __restrict__ x,
    const float* __restrict__ Wo1, const float* __restrict__ bo1,
    const float* __restrict__ Wo2, const float* __restrict__ bo2,
    const float* __restrict__ gapw, float* __restrict__ outp) {
  __shared__ float sW[64 * 32];
  __shared__ float sb[32];
  __shared__ float sw2[32];
  for (int i = threadIdx.x; i < 64 * 32; i += 256) sW[i] = Wo1[i];
  if (threadIdx.x < 32) { sb[threadIdx.x] = bo1[threadIdx.x]; sw2[threadIdx.x] = Wo2[threadIdx.x]; }
  __syncthreads();
  int n = blockIdx.x * 256 + threadIdx.x;
  if (n >= NN) return;
  float z[32];
  #pragma unroll
  for (int j = 0; j < 32; ++j) z[j] = sb[j];
  const float* hr = h + (size_t)n * HIDC;
  for (int k = 0; k < HIDC; ++k) {
    float hv = hr[k];
    #pragma unroll
    for (int j = 0; j < 32; ++j) z[j] += hv * sW[k * 32 + j];
  }
  float pred = bo2[0];
  #pragma unroll
  for (int j = 0; j < 32; ++j) pred += fmaxf(z[j], 0.f) * sw2[j];
  float gw = 1.f / (1.f + __expf(-gapw[0]));
  float f = gw * (x[(size_t)n * 32] * 100.f) + (1.f - gw) * pred * 100.f;
  outp[n] = fminf(fmaxf(f, 0.f), 100.f);
}

// ---------------------------------------------------------------------------
extern "C" void kernel_launch(void* const* d_in, const int* in_sizes, int n_in,
                              void* d_out, int out_size, void* d_ws, size_t ws_size,
                              hipStream_t stream) {
  // input order: x, edge_index, edge_attr, then params in dict order
  const float* x         = (const float*)d_in[0];
  const int*   ei        = (const int*)d_in[1];      // [2, EE]
  const float* edge_attr = (const float*)d_in[2];
  const float* W_in      = (const float*)d_in[3];
  const float* b_in      = (const float*)d_in[4];
  const float* bn_in_g   = (const float*)d_in[5];
  const float* bn_in_b   = (const float*)d_in[6];
  const float* bn_in_m   = (const float*)d_in[7];
  const float* bn_in_v   = (const float*)d_in[8];
  const float* lin_w     = (const float*)d_in[9];    // [3,64,256]
  const float* lin_ew    = (const float*)d_in[10];   // [3,4,256]
  const float* att_src   = (const float*)d_in[11];   // [3,4,64]
  const float* att_dst   = (const float*)d_in[12];
  const float* att_edge  = (const float*)d_in[13];
  const float* gat_bias  = (const float*)d_in[14];   // [3,64]
  const float* bn_g      = (const float*)d_in[15];
  const float* bn_b      = (const float*)d_in[16];
  const float* bn_m      = (const float*)d_in[17];
  const float* bn_v      = (const float*)d_in[18];
  const float* W_o1      = (const float*)d_in[19];   // [64,32]
  const float* b_o1      = (const float*)d_in[20];
  const float* W_o2      = (const float*)d_in[21];   // [32,1]
  const float* b_o2      = (const float*)d_in[22];
  const float* gapw      = (const float*)d_in[23];
  float* outp = (float*)d_out;

  const int* src = ei;
  const int* dst = ei + EE;

  // --- workspace carve (256B aligned) ---
  char* base = (char*)d_ws;
  size_t off = 0;
  auto carve = [&](size_t bytes) -> char* {
    char* p = base + off;
    off = (off + bytes + 255) & ~(size_t)255;
    return p;
  };
  float*     h      = (float*)    carve((size_t)NN * 64 * 4);
  _Float16*  h16    = (_Float16*) carve((size_t)NN * 64 * 2);
  _Float16*  x16    = (_Float16*) carve((size_t)NN * 32 * 2);
  _Float16*  xw16   = (_Float16*) carve((size_t)NN * 256 * 2);
  float*     outacc = (float*)    carve((size_t)NN * 64 * 4);
  float*     asrc   = (float*)    carve((size_t)NN * 4 * 4);
  float*     adst   = (float*)    carve((size_t)NN * 4 * 4);
  float*     mx     = (float*)    carve((size_t)NN * 4 * 4);
  float*     den    = (float*)    carve((size_t)NN * 4 * 4);
  float*     logits = (float*)    carve((size_t)E2T * 4 * 4);
  _Float16*  wt16   = (_Float16*) carve((size_t)3 * 256 * 64 * 2);
  _Float16*  winT   = (_Float16*) carve((size_t)64 * 32 * 2);
  float*     Asrc   = (float*)    carve((size_t)3 * 64 * 4 * 4);
  float*     Adst   = (float*)    carve((size_t)3 * 64 * 4 * 4);
  float*     Wae    = (float*)    carve((size_t)3 * 16 * 4);
  float*     lae    = (float*)    carve((size_t)3 * 4 * 4);
  float*     isc    = (float*)    carve((size_t)64 * 4);
  float*     ish    = (float*)    carve((size_t)64 * 4);
  float*     lsc    = (float*)    carve((size_t)3 * 64 * 4);
  float*     lsh    = (float*)    carve((size_t)3 * 64 * 4);
  (void)ws_size; (void)n_in; (void)in_sizes; (void)out_size;

  // --- prep ---
  cvt_x_kernel<<<cdiv(NN * 32, 256), 256, 0, stream>>>(x, x16);
  cvt_w_kernel<<<cdiv(3 * 256 * 64 + 64 * 32, 256), 256, 0, stream>>>(lin_w, W_in, wt16, winT);
  prep_kernel<<<1, 256, 0, stream>>>(edge_attr, lin_w, lin_ew, att_src, att_dst, att_edge,
                                     b_in, bn_in_g, bn_in_b, bn_in_m, bn_in_v,
                                     gat_bias, bn_g, bn_b, bn_m, bn_v,
                                     Wae, lae, Asrc, Adst, isc, ish, lsc, lsh);

  // --- input layer: WMMA GEMM + BN + ReLU (1 wave = 1 M-tile, 4 WMMAs) ---
  gemm_in_kernel<<<cdiv(NN / 16, 8), 256, 0, stream>>>(x16, winT, isc, ish, h, h16);

  // --- GAT layers ---
  for (int l = 0; l < 3; ++l) {
    fill_kernel<<<cdiv(NN * 4, 256), 256, 0, stream>>>(mx, -1e30f, NN * 4);
    fill_kernel<<<cdiv(NN * 4, 256), 256, 0, stream>>>(den, 0.f, NN * 4);
    fill_kernel<<<cdiv(NN * 64, 256), 256, 0, stream>>>(outacc, 0.f, NN * 64);

    // 1 wave = 1 M-tile x 4 N-tiles => 8 WMMAs, A fragments reused 4x
    gemm_xw_kernel<<<cdiv((NN / 16) * 4, 8), 256, 0, stream>>>(
        h16, wt16 + (size_t)l * 256 * 64, xw16);
    node_attn_kernel<<<cdiv(NN, 256), 256, 0, stream>>>(
        h, Asrc + (size_t)l * 256, Adst + (size_t)l * 256, asrc, adst);
    edge_p1_kernel<<<cdiv(E2T, 256), 256, 0, stream>>>(
        src, dst, edge_attr, Wae + (size_t)l * 16, lae + (size_t)l * 4,
        asrc, adst, logits, mx);
    edge_p2_kernel<<<cdiv(E2T, 256), 256, 0, stream>>>(dst, mx, logits, den);
    edge_p3_kernel<<<cdiv(E2T, 8), 256, 0, stream>>>(src, dst, xw16, logits, den, outacc);
    finalize_kernel<<<cdiv(NN * 64, 256), 256, 0, stream>>>(
        outacc, lsc + (size_t)l * 64, lsh + (size_t)l * 64, h, h16);
  }

  // --- output head ---
  head_kernel<<<cdiv(NN, 256), 256, 0, stream>>>(h, x, W_o1, b_o1, W_o2, b_o2, gapw, outp);
}
